// PointnetSAModuleMSG_57793079935587
// MI455X (gfx1250) — matrix operand order repaired and checked by reading, not compiled, not run on hardware
//
#include <hip/hip_runtime.h>

// ---------------------------------------------------------------------------
// Types for CDNA5 WMMA (wave32, 16x16x32 bf16 -> f32 accumulate)
// ---------------------------------------------------------------------------
typedef __bf16 bf16;
typedef __attribute__((ext_vector_type(16))) __bf16 v16bf;
typedef __attribute__((ext_vector_type(8)))  __bf16 v8bf;
typedef __attribute__((ext_vector_type(8)))  float  v8f;

union V16U { v16bf v; v8bf h[2]; };

__device__ __forceinline__ bf16 f2bf(float f) {
  union { float f; unsigned u; } x; x.f = f;
  unsigned r = x.u + 0x7FFFu + ((x.u >> 16) & 1u);   // round-to-nearest-even
  unsigned short h = (unsigned short)(r >> 16);
  bf16 b;
  __builtin_memcpy(&b, &h, 2);
  return b;
}

// ---------------------------------------------------------------------------
// 1) Farthest point sampling: one block per batch, dists in LDS, wave32
//    shuffle-butterfly argmax (2 barriers per step instead of a full tree).
// ---------------------------------------------------------------------------
#define FPS_T 512
#define FPS_W (FPS_T / 32)
__global__ void fps_kernel(const float* __restrict__ xyz,
                           float* __restrict__ new_xyz,   // ws copy (B,S,3)
                           float* __restrict__ out_xyz,   // d_out copy (B,S,3)
                           int N, int S) {
  __shared__ float dists[8192];
  __shared__ float rv[FPS_W];
  __shared__ int   ri[FPS_W];
  __shared__ float lp[3];
  const int b = blockIdx.x, t = threadIdx.x;
  const int lane = t & 31, wave = t >> 5;
  const float* p = xyz + (size_t)b * N * 3;
  for (int i = t; i < N; i += FPS_T) dists[i] = 1e10f;
  if (t == 0) {
    lp[0] = p[0]; lp[1] = p[1]; lp[2] = p[2];
    float* nx = new_xyz + (size_t)b * S * 3;
    float* ox = out_xyz + (size_t)b * S * 3;
    nx[0] = p[0]; nx[1] = p[1]; nx[2] = p[2];
    ox[0] = p[0]; ox[1] = p[1]; ox[2] = p[2];
  }
  __syncthreads();
  for (int s = 1; s < S; ++s) {
    const float bx = lp[0], by = lp[1], bz = lp[2];
    float best = -1.0f; int bi = 0;
    for (int i = t; i < N; i += FPS_T) {
      float dx = p[i * 3 + 0] - bx;
      float dy = p[i * 3 + 1] - by;
      float dz = p[i * 3 + 2] - bz;
      float d = dx * dx + dy * dy + dz * dz;
      float dd = fminf(dists[i], d);
      dists[i] = dd;
      if (dd > best) { best = dd; bi = i; }
    }
    // intra-wave butterfly argmax (prefer smaller index on ties)
    #pragma unroll
    for (int m = 16; m > 0; m >>= 1) {
      float ov = __shfl_xor(best, m, 32);
      int   oi = __shfl_xor(bi,   m, 32);
      if (ov > best || (ov == best && oi < bi)) { best = ov; bi = oi; }
    }
    if (lane == 0) { rv[wave] = best; ri[wave] = bi; }
    __syncthreads();
    if (wave == 0) {
      float wv = (lane < FPS_W) ? rv[lane] : -1.0f;
      int   wi = (lane < FPS_W) ? ri[lane] : 0;
      #pragma unroll
      for (int m = 16; m > 0; m >>= 1) {
        float ov = __shfl_xor(wv, m, 32);
        int   oi = __shfl_xor(wi, m, 32);
        if (ov > wv || (ov == wv && oi < wi)) { wv = ov; wi = oi; }
      }
      if (lane == 0) {
        lp[0] = p[wi * 3 + 0]; lp[1] = p[wi * 3 + 1]; lp[2] = p[wi * 3 + 2];
        float* nx = new_xyz + ((size_t)b * S + s) * 3;
        float* ox = out_xyz + ((size_t)b * S + s) * 3;
        nx[0] = lp[0]; nx[1] = lp[1]; nx[2] = lp[2];
        ox[0] = lp[0]; ox[1] = lp[1]; ox[2] = lp[2];
      }
    }
    __syncthreads();
  }
}

// ---------------------------------------------------------------------------
// 2) Ball query: first K in-radius indices in point order, padded w/ first hit.
// ---------------------------------------------------------------------------
__global__ void ball_query_kernel(const float* __restrict__ xyz,
                                  const float* __restrict__ new_xyz,
                                  int* __restrict__ idx,
                                  int N, int S, int K, float r2) {
  int gid = blockIdx.x * blockDim.x + threadIdx.x;
  if (gid >= 4 * S) return;
  int b = gid / S;
  const float* p = xyz + (size_t)b * N * 3;
  float cx = new_xyz[(size_t)gid * 3 + 0];
  float cy = new_xyz[(size_t)gid * 3 + 1];
  float cz = new_xyz[(size_t)gid * 3 + 2];
  int* o = idx + (size_t)gid * K;
  int cnt = 0, first = 0; bool has = false;
  for (int i = 0; i < N && cnt < K; ++i) {
    float dx = p[i * 3 + 0] - cx;
    float dy = p[i * 3 + 1] - cy;
    float dz = p[i * 3 + 2] - cz;
    if (dx * dx + dy * dy + dz * dz < r2) {
      if (!has) { first = i; has = true; }
      o[cnt++] = i;
    }
  }
  int pad = has ? first : 0;
  for (; cnt < K; ++cnt) o[cnt] = pad;
}

// ---------------------------------------------------------------------------
// 3) Gather grouped inputs -> bf16 A matrix (M x Kpad), channels:
//    [0..2] rel-xyz, [3..3+C) features, rest zero pad.
// ---------------------------------------------------------------------------
__global__ void gather_kernel(const float* __restrict__ xyz,
                              const float* __restrict__ feat,
                              const float* __restrict__ new_xyz,
                              const int* __restrict__ idx,
                              bf16* __restrict__ A,
                              int N, int S, int K, int C, int Kpad) {
  long long g = (long long)blockIdx.x * blockDim.x + threadIdx.x;
  long long total = (long long)4 * S * K * Kpad;
  if (g >= total) return;
  int c = (int)(g % Kpad);
  long long row = g / Kpad;
  int k = (int)(row % K);
  long long bs = row / K;               // b*S + s
  int b = (int)(bs / S);
  int pt = idx[bs * K + k];
  float v;
  if (c < 3)
    v = xyz[((size_t)b * N + pt) * 3 + c] - new_xyz[bs * 3 + c];
  else if (c < 3 + C)
    v = feat[((size_t)b * N + pt) * C + (c - 3)];
  else
    v = 0.0f;
  A[(size_t)row * Kpad + c] = f2bf(v);
}

// ---------------------------------------------------------------------------
// 4) Weight prep: fp32 W (cin x cout) -> bf16 Wt (cout x Kpad), zero padded.
// ---------------------------------------------------------------------------
__global__ void prep_w(const float* __restrict__ W, bf16* __restrict__ Wt,
                       int cin, int cout, int Kpad) {
  int g = blockIdx.x * blockDim.x + threadIdx.x;
  if (g >= cout * Kpad) return;
  int k = g % Kpad, n = g / Kpad;
  Wt[g] = f2bf(k < cin ? W[(size_t)k * cout + n] : 0.0f);
}

__global__ void zero_f32(float* p, int n) {
  int g = blockIdx.x * blockDim.x + threadIdx.x;
  if (g < n) p[g] = 0.0f;
}

// ---------------------------------------------------------------------------
// 5) WMMA GEMM, templated on (K, COUT) so K-chunks and N-tiles fully unroll.
//    One wave per 16-row M-stripe: A (16 x K) is loaded into registers ONCE
//    (K<=96 -> <=24 VGPRs/lane) and reused across all COUT/16 output tiles,
//    cutting A traffic by COUT/16 (8x for the 128-wide layer).
//    Register layouts per ISA 7.12.2:
//      A: lanes 0-15 row=l, K chunks {0..7,16..23}; lanes 16-31 {8..15,24..31}
//      B: Wt is (COUT x K); lanes 0-15 col=l K {0..7,16..23}; lanes 16-31 {8..15,24..31}
//      C: VGPR r -> row r (lanes 0-15) / row r+8 (lanes 16-31), col = lane&15
//    Per-channel sum/sumsq accumulated with f32 global atomics for BN.
// ---------------------------------------------------------------------------
template <int K, int COUT>
__global__ void wmma_gemm_stats(const bf16* __restrict__ A,
                                const bf16* __restrict__ Wt,
                                float* __restrict__ Cbuf,
                                float* __restrict__ stats,   // [2*COUT]
                                int M) {
  constexpr int KC = K / 32;
  constexpr int TN = COUT / 16;
  int tileM = (blockIdx.x * blockDim.x + threadIdx.x) >> 5;
  int lane  = threadIdx.x & 31;
  if (tileM * 16 >= M) return;          // whole wave exits together (EXEC all-1s)
  int half = lane >> 4;
  int l    = lane & 15;

  // preload A stripe into registers
  const bf16* arow = A + (size_t)(tileM * 16 + l) * K + half * 8;
  V16U a[KC];
  #pragma unroll
  for (int kc = 0; kc < KC; ++kc) {
    a[kc].h[0] = *(const v8bf*)(arow + kc * 32);       // K = kc*32 + half*8 + 0..7
    a[kc].h[1] = *(const v8bf*)(arow + kc * 32 + 16);  // K = kc*32 + 16 + half*8 + 0..7
  }

  float* crow0 = Cbuf + (size_t)(tileM * 16 + half * 8) * COUT;

  #pragma unroll
  for (int tn = 0; tn < TN; ++tn) {
    const bf16* brow = Wt + (size_t)(tn * 16 + l) * K + half * 8;
    v8f acc = {};
    #pragma unroll
    for (int kc = 0; kc < KC; ++kc) {
      V16U bm;
      bm.h[0] = *(const v8bf*)(brow + kc * 32);        // K = kc*32 + half*8 + 0..7
      bm.h[1] = *(const v8bf*)(brow + kc * 32 + 16);   // K = kc*32 + 16 + half*8 + 0..7
      acc = __builtin_amdgcn_wmma_f32_16x16x32_bf16(
          /*neg_a=*/false, a[kc].v, /*neg_b=*/false, bm.v,
          /*c_mod=*/(short)0, acc, /*reuse_a=*/false, /*reuse_b=*/false);
    }
    int col = tn * 16 + l;
    float s = 0.0f, sq = 0.0f;
    float* crow = crow0 + col;
    #pragma unroll
    for (int r = 0; r < 8; ++r) {
      float v = acc[r];
      crow[(size_t)r * COUT] = v;
      s += v; sq += v * v;
    }
    atomicAdd(&stats[col], s);
    atomicAdd(&stats[COUT + col], sq);
  }
}

// ---------------------------------------------------------------------------
// 6) BN finalize: per-channel mean/var -> scale/bias.
// ---------------------------------------------------------------------------
__global__ void bn_finalize(const float* __restrict__ stats,
                            const float* __restrict__ gamma,
                            const float* __restrict__ beta,
                            float* __restrict__ sb, int Cout, float invM) {
  int c = threadIdx.x;
  if (c >= Cout) return;
  float mean = stats[c] * invM;
  float var  = stats[Cout + c] * invM - mean * mean;
  float sc   = gamma[c] * rsqrtf(var + 1e-5f);
  sb[c]        = sc;
  sb[Cout + c] = beta[c] - mean * sc;
}

// ---------------------------------------------------------------------------
// 7a) BN + ReLU -> bf16 input of next layer (Kpad_next == Cout here).
// ---------------------------------------------------------------------------
__global__ void bn_relu_bf16(const float* __restrict__ Cbuf,
                             const float* __restrict__ sb,
                             bf16* __restrict__ Anext,
                             long long total, int Cout) {
  long long g = (long long)blockIdx.x * blockDim.x + threadIdx.x;
  if (g >= total) return;
  int c = (int)(g % Cout);
  float v = Cbuf[g] * sb[c] + sb[Cout + c];
  Anext[g] = f2bf(fmaxf(v, 0.0f));
}

// ---------------------------------------------------------------------------
// 7b) Final layer: BN + ReLU + max over nsample -> output feature columns.
// ---------------------------------------------------------------------------
__global__ void bn_relu_maxpool(const float* __restrict__ Cbuf,
                                const float* __restrict__ sb,
                                float* __restrict__ out,
                                int BS, int K, int Cout,
                                int outStride, int colOff) {
  int g = blockIdx.x * blockDim.x + threadIdx.x;
  if (g >= BS * Cout) return;
  int c = g % Cout, bs = g / Cout;
  float sc = sb[c], bi = sb[Cout + c];
  float m = 0.0f;                        // ReLU => max >= 0
  for (int k = 0; k < K; ++k) {
    float v = Cbuf[((size_t)bs * K + k) * Cout + c] * sc + bi;
    m = fmaxf(m, fmaxf(v, 0.0f));
  }
  out[(size_t)bs * outStride + colOff + c] = m;
}

// ---------------------------------------------------------------------------
// GEMM dispatch table over the six (Kpad, Cout) shapes in this module.
// ---------------------------------------------------------------------------
static void launch_gemm(int kp, int cout, const bf16* A, const bf16* Wt,
                        float* Cbuf, float* stats, int M, hipStream_t stream) {
  dim3 grid((M / 16 + 7) / 8), block(256);
  if      (kp == 96 && cout == 32)
    wmma_gemm_stats<96, 32><<<grid, block, 0, stream>>>(A, Wt, Cbuf, stats, M);
  else if (kp == 32 && cout == 32)
    wmma_gemm_stats<32, 32><<<grid, block, 0, stream>>>(A, Wt, Cbuf, stats, M);
  else if (kp == 32 && cout == 64)
    wmma_gemm_stats<32, 64><<<grid, block, 0, stream>>>(A, Wt, Cbuf, stats, M);
  else if (kp == 96 && cout == 64)
    wmma_gemm_stats<96, 64><<<grid, block, 0, stream>>>(A, Wt, Cbuf, stats, M);
  else if (kp == 64 && cout == 64)
    wmma_gemm_stats<64, 64><<<grid, block, 0, stream>>>(A, Wt, Cbuf, stats, M);
  else /* kp == 64 && cout == 128 */
    wmma_gemm_stats<64, 128><<<grid, block, 0, stream>>>(A, Wt, Cbuf, stats, M);
}

// ---------------------------------------------------------------------------
// Host orchestration (graph-capture safe: kernels only, on `stream`).
// Input layout (setup_inputs insertion order):
//   [0]=xyz (4,8192,3) f32, [1]=features (4,8192,64) f32,
//   then per layer L = branch*3 + layer: [2+3L]=W, [3+3L]=gamma, [4+3L]=beta.
// Output: new_xyz (4*2048*3 f32) then new_features (4*2048*192 f32).
// ---------------------------------------------------------------------------
extern "C" void kernel_launch(void* const* d_in, const int* in_sizes, int n_in,
                              void* d_out, int out_size, void* d_ws, size_t ws_size,
                              hipStream_t stream) {
  (void)in_sizes; (void)n_in; (void)out_size; (void)ws_size;
  const int B = 4, N = 8192, C = 64, S = 2048, BS = B * S;
  const int KPAD0 = 96;                                    // pad 67 -> 96 (3x32)
  const int   nsamp[2] = {16, 32};
  const float r2s[2]   = {0.25f, 1.0f};
  const int   dims[2][4] = {{67, 32, 32, 64}, {67, 64, 64, 128}};

  const float* xyz  = (const float*)d_in[0];
  const float* feat = (const float*)d_in[1];

  // workspace carve-out
  char* base = (char*)d_ws;
  size_t off = 0;
  auto take = [&](size_t bytes) -> void* {
    void* p = base + off;
    off = (off + bytes + 255) & ~(size_t)255;
    return p;
  };
  float* nx_ws  = (float*)take((size_t)BS * 3 * 4);
  int*   idxbuf = (int*)  take((size_t)BS * 32 * 4);
  bf16*  Wt[6];
  for (int i = 0; i < 6; ++i) Wt[i] = (bf16*)take((size_t)128 * 96 * 2);
  float* stats = (float*)take(256 * 4);
  float* sb    = (float*)take(256 * 4);
  const size_t MMAX = (size_t)BS * 32;                     // 262144
  bf16*  bufA = (bf16*) take(MMAX * 128 * 2);
  bf16*  bufB = (bf16*) take(MMAX * 128 * 2);
  float* Cbuf = (float*)take(MMAX * 128 * 4);

  // pre-transpose + pad weights to bf16 (Cout x Kpad)
  for (int br = 0; br < 2; ++br) {
    for (int l = 0; l < 3; ++l) {
      int L = br * 3 + l;
      int cin = dims[br][l], cout = dims[br][l + 1];
      int kp = (l == 0) ? KPAD0 : cin;
      const float* W = (const float*)d_in[2 + 3 * L + 0];
      int n = cout * kp;
      prep_w<<<(n + 255) / 256, 256, 0, stream>>>(W, Wt[L], cin, cout, kp);
    }
  }

  float* out_xyz  = (float*)d_out;
  float* out_feat = out_xyz + (size_t)BS * 3;

  fps_kernel<<<B, FPS_T, 0, stream>>>(xyz, nx_ws, out_xyz, N, S);

  int colOff = 0;
  for (int br = 0; br < 2; ++br) {
    const int K = nsamp[br];
    const int M = BS * K;

    ball_query_kernel<<<(BS + 255) / 256, 256, 0, stream>>>(
        xyz, nx_ws, idxbuf, N, S, K, r2s[br]);

    long long tg = (long long)M * KPAD0;
    gather_kernel<<<(unsigned)((tg + 255) / 256), 256, 0, stream>>>(
        xyz, feat, nx_ws, idxbuf, bufA, N, S, K, C, KPAD0);

    bf16* Ain = bufA;
    bf16* Aout = bufB;
    for (int l = 0; l < 3; ++l) {
      int L = br * 3 + l;
      int cout = dims[br][l + 1];
      int kp = (l == 0) ? KPAD0 : dims[br][l];
      const float* gamma = (const float*)d_in[2 + 3 * L + 1];
      const float* beta  = (const float*)d_in[2 + 3 * L + 2];

      zero_f32<<<1, 256, 0, stream>>>(stats, 2 * cout);

      launch_gemm(kp, cout, Ain, Wt[L], Cbuf, stats, M, stream);

      bn_finalize<<<1, 128, 0, stream>>>(stats, gamma, beta, sb, cout,
                                         1.0f / (float)M);

      if (l < 2) {
        long long n = (long long)M * cout;
        bn_relu_bf16<<<(unsigned)((n + 255) / 256), 256, 0, stream>>>(
            Cbuf, sb, Aout, n, cout);
        bf16* t = Ain; Ain = Aout; Aout = t;
      } else {
        int n = BS * cout;
        bn_relu_maxpool<<<(n + 255) / 256, 256, 0, stream>>>(
            Cbuf, sb, out_feat, BS, K, cout, 192, colOff);
      }
    }
    colOff += dims[br][3];
  }
}
